// Attention_2851858284851
// MI455X (gfx1250) — matrix operand order repaired and checked by reading, not compiled, not run on hardware
//
#include <hip/hip_runtime.h>
#include <hip/hip_bf16.h>
#include <stdint.h>

// ---------------------------------------------------------------------------
// Problem constants (match reference)
// ---------------------------------------------------------------------------
#define L_SEQ   2048
#define HIDDEN  2048
#define DH      128
#define NH      32
#define NKV     4
#define GQA_G   8          // NH / NKV
#define EPS_F   1e-6f
#define INV_SQRT_D 0.08838834764831845f   // 1/sqrt(128)

// ---------------------------------------------------------------------------
// WMMA types (CDNA5 wave32): v16bf A/B fragments, v8f C/D accumulator
// ---------------------------------------------------------------------------
typedef __attribute__((ext_vector_type(16))) __bf16       v16bf;
typedef __attribute__((ext_vector_type(8)))  float        v8f;
typedef __attribute__((ext_vector_type(4)))  unsigned int v4u;

union Frag16 { v16bf v; v4u q[2]; };

__device__ __forceinline__ unsigned short f2bf(float f) {
  union { float f; unsigned int u; } x; x.f = f;
  unsigned int r = x.u + 0x7FFFu + ((x.u >> 16) & 1u);   // RNE
  return (unsigned short)(r >> 16);
}

// ---------------------------------------------------------------------------
// f32 -> bf16 conversion (grid-stride)
// ---------------------------------------------------------------------------
__global__ void cvt_f32_bf16(unsigned short* __restrict__ dst,
                             const float* __restrict__ src, size_t n) {
  size_t i = (size_t)blockIdx.x * blockDim.x + threadIdx.x;
  size_t stride = (size_t)gridDim.x * blockDim.x;
  for (; i < n; i += stride) dst[i] = f2bf(src[i]);
}

// ---------------------------------------------------------------------------
// GEMM: C[M,N] (f32) = A[M,K] (bf16, row-major) x B[K,N], with B supplied as
// BT[N,K] row-major (== original weight layout (out_features, in_features)).
//
// Block = 128 threads = 4 waves. Each wave computes a 32(M) x 64(N) tile of C
// (8 f32 accumulators); the block covers 128x64. Per K-step a wave issues
// 12 x b128 loads (2 A frags + 4 B frags) feeding 8 WMMA: each B fragment is
// reused by two WMMAs, keeping VMEM issue rate at 1.5 loads/WMMA so the
// matrix pipes, not the vector-memory port, set the pace.
//
// Per the CDNA5 ISA wave32 layouts:
//   A 16x32 bf16: lane m = lane&15; halves[0..7]=K[kb..kb+7],
//                 halves[8..15]=K[16+kb..16+kb+7], kb = (lane<16)?0:8
//   B 32x16 bf16: lane n = lane&15; same K order as A (kb2=(lane<16)?0:8)
//   C 16x16 f32 : c[r] = C[r + 8*(lane>=16)][lane&15]
// ---------------------------------------------------------------------------
__global__ void gemm_bf16_wmma(float* __restrict__ C,
                               const unsigned short* __restrict__ A,
                               const unsigned short* __restrict__ BT,
                               int M, int N, int K) {
  const int lane = threadIdx.x & 31;
  const int wave = threadIdx.x >> 5;
  const int m0 = blockIdx.y * 128 + wave * 32;
  const int n0 = blockIdx.x * 64;
  const int nlo = lane & 15;
  const int hi  = lane >> 4;

  const v8f vzero = {0.f,0.f,0.f,0.f,0.f,0.f,0.f,0.f};
  v8f acc[2][4];
#pragma unroll
  for (int g = 0; g < 2; ++g)
#pragma unroll
    for (int j = 0; j < 4; ++j) acc[g][j] = vzero;

  const int akb  = hi * 8;    // A fragment K sub-base (halves)
  const int bkb  = hi * 8;    // B fragment K sub-base (halves)
  const unsigned short* arow0 = A + (size_t)(m0 + nlo) * K;
  const unsigned short* arow1 = arow0 + (size_t)16 * K;
  const unsigned short* br[4];
#pragma unroll
  for (int j = 0; j < 4; ++j)
    br[j] = BT + (size_t)(n0 + j * 16 + nlo) * K + bkb;

  for (int k0 = 0; k0 < K; k0 += 32) {
    __builtin_prefetch(arow0 + k0 + 256, 0, 3);   // WGP-scope prefetch of next A lines
    Frag16 af0, af1;
    af0.q[0] = *(const v4u*)(arow0 + k0 + akb);
    af0.q[1] = *(const v4u*)(arow0 + k0 + 16 + akb);
    af1.q[0] = *(const v4u*)(arow1 + k0 + akb);
    af1.q[1] = *(const v4u*)(arow1 + k0 + 16 + akb);

#pragma unroll
    for (int j = 0; j < 4; ++j) {
      Frag16 bf;
      bf.q[0] = *(const v4u*)(br[j] + k0);
      bf.q[1] = *(const v4u*)(br[j] + k0 + 16);
      acc[0][j] = __builtin_amdgcn_wmma_f32_16x16x32_bf16(false, af0.v, false, bf.v, (short)0, acc[0][j], false, false);
      acc[1][j] = __builtin_amdgcn_wmma_f32_16x16x32_bf16(false, af1.v, false, bf.v, (short)0, acc[1][j], false, false);
    }
  }

#pragma unroll
  for (int g = 0; g < 2; ++g) {
    float* crow = C + (size_t)(m0 + g * 16 + hi * 8) * N + n0 + nlo;
#pragma unroll
    for (int r = 0; r < 8; ++r) {
      crow[(size_t)r * N +  0] = acc[g][0][r];
      crow[(size_t)r * N + 16] = acc[g][1][r];
      crow[(size_t)r * N + 32] = acc[g][2][r];
      crow[(size_t)r * N + 48] = acc[g][3][r];
    }
  }
}

// ---------------------------------------------------------------------------
// Per-(token, head) RMSNorm + RoPE; writes head-major bf16 [h][l][d].
// outscale folds 1/sqrt(D) into Q. grid = (L, nheads), block = 128 (= D).
// ---------------------------------------------------------------------------
__global__ void norm_rope_bf16(unsigned short* __restrict__ dst,
                               const float* __restrict__ src,
                               const float* __restrict__ w,
                               const float* __restrict__ cosp,
                               const float* __restrict__ sinp,
                               int nheads, float outscale) {
  const int l  = blockIdx.x;
  const int hh = blockIdx.y;
  const int d  = threadIdx.x;

  const float val = src[((size_t)l * nheads + hh) * DH + d];

  __shared__ float red[DH];
  __shared__ float nv[DH];
  red[d] = val * val;
  __syncthreads();
#pragma unroll
  for (int s = DH / 2; s > 0; s >>= 1) {
    if (d < s) red[d] += red[d + s];
    __syncthreads();
  }
  const float rs  = rsqrtf(red[0] * (1.0f / DH) + EPS_F);
  const float nrm = val * rs * w[d];
  nv[d] = nrm;
  __syncthreads();

  const float part = (d < DH / 2) ? -nv[d + DH / 2] : nv[d - DH / 2];
  const float c = cosp[(size_t)l * DH + d];
  const float s = sinp[(size_t)l * DH + d];
  const float o = (nrm * c + part * s) * outscale;
  dst[((size_t)hh * L_SEQ + l) * DH + d] = f2bf(o);
}

// ---------------------------------------------------------------------------
// V transpose + convert: vt[kvh][d][l] = bf16(v_f32[l][kvh*D + d])
// so P x V WMMA B-fragments are contiguous 128-bit loads.
// ---------------------------------------------------------------------------
__global__ void v_transpose_bf16(unsigned short* __restrict__ vt,
                                 const float* __restrict__ v, size_t n) {
  size_t i = (size_t)blockIdx.x * blockDim.x + threadIdx.x;
  size_t stride = (size_t)gridDim.x * blockDim.x;
  for (; i < n; i += stride) {
    int l = (int)(i % L_SEQ);
    size_t t = i / L_SEQ;
    int d   = (int)(t % DH);
    int kvh = (int)(t / DH);
    vt[i] = f2bf(v[((size_t)l * NKV + kvh) * DH + d]);
  }
}

// ---------------------------------------------------------------------------
// Flash attention, one wave per 16-row Q tile per head. Q pre-scaled by
// 1/sqrt(D). Streams 32 keys/step: 8 WMMA for S, online softmax with
// __shfl_xor row reductions (width-16 groups match C-layout striping),
// P staged through LDS (C-layout -> A-layout), 8 WMMA for P*V.
// grid = (L/16/4, NH), block = 128 (4 waves).
// ---------------------------------------------------------------------------
__global__ void flash_attn_wmma(unsigned short* __restrict__ attn,      // [L][NH*DH] bf16
                                const unsigned short* __restrict__ qbf, // [NH][L][DH]
                                const unsigned short* __restrict__ kbf, // [NKV][L][DH]
                                const unsigned short* __restrict__ vtbf) { // [NKV][DH][L]
  const int lane = threadIdx.x & 31;
  const int wave = threadIdx.x >> 5;
  const int qt   = blockIdx.x * 4 + wave;
  const int m0   = qt * 16;
  const int h    = blockIdx.y;
  const int kvh  = h / GQA_G;
  const int nlo  = lane & 15;
  const int hi   = lane >> 4;

  const unsigned short* qh = qbf + ((size_t)h   * L_SEQ + m0) * DH;
  const unsigned short* kh = kbf +  (size_t)kvh * L_SEQ * DH;
  const unsigned short* vh = vtbf + (size_t)kvh * DH * L_SEQ;

  // Resident Q fragments: 4 chunks of D (16x32 each), A-layout.
  Frag16 qf[4];
  {
    const unsigned short* qrow = qh + (size_t)nlo * DH;
#pragma unroll
    for (int t = 0; t < 4; ++t) {
      const int base = t * 32 + hi * 8;
      qf[t].q[0] = *(const v4u*)(qrow + base);
      qf[t].q[1] = *(const v4u*)(qrow + base + 16);
    }
  }

  const v8f vzero = {0.f,0.f,0.f,0.f,0.f,0.f,0.f,0.f};
  v8f o[8];
#pragma unroll
  for (int dt = 0; dt < 8; ++dt) o[dt] = vzero;
  float mr[8], lr[8];
#pragma unroll
  for (int r = 0; r < 8; ++r) { mr[r] = -1e30f; lr[r] = 0.f; }

  __shared__ unsigned short pb[4][16][32];   // per-wave P staging (bf16)
  unsigned short (*P)[32] = pb[wave];

  const int qmax = m0 + 15;
  for (int j0 = 0; j0 <= qmax; j0 += 32) {
    // ---- S = Q * K^T for 32 keys (two 16-wide column tiles) ----
    v8f s0 = vzero, s1 = vzero;
#pragma unroll
    for (int t = 0; t < 4; ++t) {
      const int dbase = t * 32 + hi * 8;
      const unsigned short* kr0 = kh + (size_t)(j0 + nlo)      * DH + dbase;
      const unsigned short* kr1 = kh + (size_t)(j0 + 16 + nlo) * DH + dbase;
      Frag16 b0, b1;
      b0.q[0] = *(const v4u*)(kr0); b0.q[1] = *(const v4u*)(kr0 + 16);
      b1.q[0] = *(const v4u*)(kr1); b1.q[1] = *(const v4u*)(kr1 + 16);
      s0 = __builtin_amdgcn_wmma_f32_16x16x32_bf16(false, qf[t].v, false, b0.v, (short)0, s0, false, false);
      s1 = __builtin_amdgcn_wmma_f32_16x16x32_bf16(false, qf[t].v, false, b1.v, (short)0, s1, false, false);
    }

    // ---- causal mask + online softmax over this 32-key slab ----
    const int k0pos = j0 + nlo;
    const int k1pos = j0 + 16 + nlo;
#pragma unroll
    for (int r = 0; r < 8; ++r) {
      const int qr = m0 + r + hi * 8;
      if (k0pos > qr) s0[r] = -1e30f;
      if (k1pos > qr) s1[r] = -1e30f;

      float tmax = fmaxf(s0[r], s1[r]);
#pragma unroll
      for (int off = 1; off < 16; off <<= 1)
        tmax = fmaxf(tmax, __shfl_xor(tmax, off));

      const float mnew = fmaxf(mr[r], tmax);
      const float sc   = __expf(mr[r] - mnew);
      mr[r] = mnew;

      const float p0 = __expf(s0[r] - mnew);
      const float p1 = __expf(s1[r] - mnew);
      float rsum = p0 + p1;
#pragma unroll
      for (int off = 1; off < 16; off <<= 1)
        rsum += __shfl_xor(rsum, off);
      lr[r] = lr[r] * sc + rsum;

      // rescale running output row r
#pragma unroll
      for (int dt = 0; dt < 8; ++dt) o[dt][r] *= sc;

      // stage P (C-layout element -> LDS [row][key])
      P[r + hi * 8][nlo]      = f2bf(p0);
      P[r + hi * 8][16 + nlo] = f2bf(p1);
    }

    // LDS RAW within the same wave: wait on DScnt before A-layout reload.
    asm volatile("s_wait_dscnt 0" ::: "memory");

    // ---- reload P in A-layout, accumulate O += P * V ----
    Frag16 pf;
    {
      const unsigned short* prow = &P[nlo][0];
      const int kb = hi * 8;
      pf.q[0] = *(const v4u*)(prow + kb);
      pf.q[1] = *(const v4u*)(prow + 16 + kb);
    }
#pragma unroll
    for (int dt = 0; dt < 8; ++dt) {
      const unsigned short* vr = vh + (size_t)(dt * 16 + nlo) * L_SEQ + j0 + hi * 8;
      Frag16 bv;
      bv.q[0] = *(const v4u*)(vr);
      bv.q[1] = *(const v4u*)(vr + 16);
      o[dt] = __builtin_amdgcn_wmma_f32_16x16x32_bf16(false, pf.v, false, bv.v, (short)0, o[dt], false, false);
    }
  }

  // ---- epilogue: O /= l, store bf16 row-major [l][h*D + d] ----
#pragma unroll
  for (int r = 0; r < 8; ++r) {
    const float inv_l = 1.0f / lr[r];
    const size_t row = (size_t)(m0 + r + hi * 8) * (NH * DH) + (size_t)h * DH + nlo;
#pragma unroll
    for (int dt = 0; dt < 8; ++dt)
      attn[row + dt * 16] = f2bf(o[dt][r] * inv_l);
  }
}

// ---------------------------------------------------------------------------
// Host-side orchestration
// ---------------------------------------------------------------------------
extern "C" void kernel_launch(void* const* d_in, const int* in_sizes, int n_in,
                              void* d_out, int out_size, void* d_ws, size_t ws_size,
                              hipStream_t stream) {
  const float* x    = (const float*)d_in[0];
  const float* wq   = (const float*)d_in[1];
  const float* wk   = (const float*)d_in[2];
  const float* wv   = (const float*)d_in[3];
  const float* wo   = (const float*)d_in[4];
  const float* qw   = (const float*)d_in[5];
  const float* kw   = (const float*)d_in[6];
  const float* cosp = (const float*)d_in[7];
  const float* sinp = (const float*)d_in[8];
  float* out = (float*)d_out;

  char* ws = (char*)d_ws;
  size_t off = 0;
  auto alloc = [&](size_t bytes) -> void* {
    void* p = ws + off;
    off = (off + bytes + 255) & ~(size_t)255;
    return p;
  };

  unsigned short* xbf   = (unsigned short*)alloc((size_t)L_SEQ * HIDDEN * 2);
  unsigned short* wqbf  = (unsigned short*)alloc((size_t)NH  * DH * HIDDEN * 2);
  unsigned short* wkbf  = (unsigned short*)alloc((size_t)NKV * DH * HIDDEN * 2);
  unsigned short* wvbf  = (unsigned short*)alloc((size_t)NKV * DH * HIDDEN * 2);
  unsigned short* wobf  = (unsigned short*)alloc((size_t)HIDDEN * NH * DH * 2);
  float*          qf32  = (float*)alloc((size_t)L_SEQ * NH  * DH * 4);
  float*          kf32  = (float*)alloc((size_t)L_SEQ * NKV * DH * 4);
  float*          vf32  = (float*)alloc((size_t)L_SEQ * NKV * DH * 4);
  unsigned short* qbf   = (unsigned short*)alloc((size_t)NH  * L_SEQ * DH * 2);
  unsigned short* kbf   = (unsigned short*)alloc((size_t)NKV * L_SEQ * DH * 2);
  unsigned short* vtbf  = (unsigned short*)alloc((size_t)NKV * DH * L_SEQ * 2);
  unsigned short* attnb = (unsigned short*)alloc((size_t)L_SEQ * NH * DH * 2);

  // 1) bf16 conversion of activations + weights (weights keep (O,K) layout == BT)
  cvt_f32_bf16<<<2048, 256, 0, stream>>>(xbf,  x,  (size_t)L_SEQ * HIDDEN);
  cvt_f32_bf16<<<2048, 256, 0, stream>>>(wqbf, wq, (size_t)NH  * DH * HIDDEN);
  cvt_f32_bf16<<<1024, 256, 0, stream>>>(wkbf, wk, (size_t)NKV * DH * HIDDEN);
  cvt_f32_bf16<<<1024, 256, 0, stream>>>(wvbf, wv, (size_t)NKV * DH * HIDDEN);
  cvt_f32_bf16<<<2048, 256, 0, stream>>>(wobf, wo, (size_t)HIDDEN * NH * DH);

  // 2) QKV projections (WMMA bf16 -> f32); block covers 128(M) x 64(N)
  gemm_bf16_wmma<<<dim3((NH * DH) / 64, L_SEQ / 128), 128, 0, stream>>>(
      qf32, xbf, wqbf, L_SEQ, NH * DH, HIDDEN);
  gemm_bf16_wmma<<<dim3((NKV * DH) / 64, L_SEQ / 128), 128, 0, stream>>>(
      kf32, xbf, wkbf, L_SEQ, NKV * DH, HIDDEN);
  gemm_bf16_wmma<<<dim3((NKV * DH) / 64, L_SEQ / 128), 128, 0, stream>>>(
      vf32, xbf, wvbf, L_SEQ, NKV * DH, HIDDEN);

  // 3) RMSNorm + RoPE (fold 1/sqrt(D) into Q), V transpose
  norm_rope_bf16<<<dim3(L_SEQ, NH),  DH, 0, stream>>>(qbf, qf32, qw, cosp, sinp, NH,  INV_SQRT_D);
  norm_rope_bf16<<<dim3(L_SEQ, NKV), DH, 0, stream>>>(kbf, kf32, kw, cosp, sinp, NKV, 1.0f);
  v_transpose_bf16<<<1024, 256, 0, stream>>>(vtbf, vf32, (size_t)NKV * DH * L_SEQ);

  // 4) Causal GQA flash attention (WMMA), one wave per 16-row Q tile
  flash_attn_wmma<<<dim3(L_SEQ / 16 / 4, NH), 128, 0, stream>>>(attnb, qbf, kbf, vtbf);

  // 5) Output projection -> f32 result
  gemm_bf16_wmma<<<dim3(HIDDEN / 64, L_SEQ / 128), 128, 0, stream>>>(
      out, attnb, wobf, L_SEQ, HIDDEN, NH * DH);
}
